// CausalMultiHeadAttention_55671366091375
// MI455X (gfx1250) — matrix operand order repaired and checked by reading, not compiled, not run on hardware
//
#include <hip/hip_runtime.h>

typedef __attribute__((ext_vector_type(16))) _Float16 v16h;
typedef __attribute__((ext_vector_type(8)))  float    v8f;
typedef __attribute__((ext_vector_type(4)))  unsigned int v4u;
typedef __attribute__((ext_vector_type(8)))  int      v8i;
typedef __attribute__((ext_vector_type(4)))  int      v4i;

#define D_MODEL   1024
#define NUM_HEADS 16
#define DK        64
#define SEQ       2048
#define BATCH     2

#if defined(__has_include)
#if __has_include(<hip/amd_detail/amd_gfx1250_TDM.h>)
#define TDM_6ARG 1
#endif
#endif

static __device__ inline v8f wmma_f16(v16h a, v16h b, v8f c) {
  // D = A(16x32 f16) * B(32x16 f16) + C(16x16 f32)
  return __builtin_amdgcn_wmma_f32_16x16x32_f16(false, a, false, b, (short)0, c,
                                                false, false);
}

// ---- fragment loaders (contiguous-k layouts, wave32) -------------------------
// A 16x32 (MxK): lanes 0-15 row=lane, halfs0-15=K0..15;
//                lanes 16-31 row=lane-16, halfs0-15=K16..31.
static __device__ inline v16h load_frag_a_h(const _Float16* p, int row0, int ld, int k0) {
  const int lane = threadIdx.x & 31;
  const _Float16* r = p + (size_t)(row0 + (lane & 15)) * ld + k0 + ((lane & 16) ? 16 : 0);
  v16h a;
#pragma unroll
  for (int i = 0; i < 8; ++i) { a[i] = r[i]; a[8 + i] = r[8 + i]; }
  return a;
}
// B 32x16 (KxN) sourced from a row-major [N][K] tile (element(k,n) = p[n*ld+k]):
// lanes 0-15 col=lane, K=0..15 contiguous; lanes 16-31 col=lane-16, K=16..31.
static __device__ inline v16h load_frag_bt(const _Float16* p, int col0, int ld, int k0) {
  const int lane = threadIdx.x & 31;
  const _Float16* r = p + (size_t)(col0 + (lane & 15)) * ld + k0 + ((lane & 16) ? 16 : 0);
  v16h b;
#pragma unroll
  for (int i = 0; i < 16; ++i) b[i] = r[i];
  return b;
}

// ---- Tensor Data Mover: 2D f16 tile load, global -> LDS (ISA ch.8 D#) -----
static __device__ inline void tdm_load_2d_f16(unsigned lds_byte_addr,
                                              const void* gptr,
                                              unsigned tensor_d0, unsigned tensor_d1,
                                              unsigned tile_d0, unsigned tile_d1,
                                              unsigned long long stride0_elems,
                                              unsigned pad_interval_code,
                                              unsigned pad_amount_code,
                                              unsigned pad_enable) {
  unsigned long long ga = (unsigned long long)gptr;
  v4u g0;
  g0[0] = 1u;                                            // count=1, user descriptor
  g0[1] = lds_byte_addr;                                 // D#.lds_addr
  g0[2] = (unsigned)(ga & 0xffffffffu);                  // global_addr[31:0]
  g0[3] = (unsigned)((ga >> 32) & 0x01ffffffu) | 0x80000000u;  // [56:32] | type=2
  v8i g1;
  g1[0] = (int)((1u << 16)                               // data_size = 2 bytes
                | ((pad_enable & 1u) << 20)
                | ((pad_interval_code & 7u) << 22)
                | ((pad_amount_code & 0x7fu) << 25));
  g1[1] = (int)((tensor_d0 & 0xffffu) << 16);            // tensor_dim0 @ bit48
  g1[2] = (int)(((tensor_d0 >> 16) & 0xffffu) | ((tensor_d1 & 0xffffu) << 16));
  g1[3] = (int)(((tensor_d1 >> 16) & 0xffffu) | ((tile_d0 & 0xffffu) << 16));
  g1[4] = (int)(tile_d1 & 0xffffu);                      // tile_dim1 (tile_dim2=0)
  g1[5] = (int)(unsigned)(stride0_elems & 0xffffffffu);  // tensor_dim0_stride[31:0]
  g1[6] = (int)(unsigned)((stride0_elems >> 32) & 0xffffu);
  g1[7] = 0;
  v4i z4 = {0, 0, 0, 0};
#ifdef TDM_6ARG
  v8i z8 = {0, 0, 0, 0, 0, 0, 0, 0};
  __builtin_amdgcn_tensor_load_to_lds(g0, g1, z4, z4, z8, 0);
#else
  __builtin_amdgcn_tensor_load_to_lds(g0, g1, z4, z4, 0);
#endif
}

// ---------------------------------------------------------------------------
// Kernel 1: QKV projection.  out[b,h,s,d] = (x @ W + bias), f16 head layout.
// Double-buffered LDS; B fragments hoisted ahead of the WMMA chain.
__global__ __launch_bounds__(128) void qkv_proj_kernel(
    const float* __restrict__ x,
    const float* __restrict__ Wq, const float* __restrict__ bq,
    const float* __restrict__ Wk, const float* __restrict__ bk,
    const float* __restrict__ Wv, const float* __restrict__ bv,
    _Float16* __restrict__ Qh, _Float16* __restrict__ Kh, _Float16* __restrict__ Vh) {
  const int tid = threadIdx.x, lane = tid & 31, wave = tid >> 5;
  const int m0  = blockIdx.x * 64;                 // flat row in [0, B*S)
  const int mat = blockIdx.y >> 4;                 // 0=Q 1=K 2=V
  const int nt  = (blockIdx.y & 15) * 64;          // col offset in D_MODEL
  const float*    W    = (mat == 0) ? Wq : (mat == 1) ? Wk : Wv;
  const float*    bias = (mat == 0) ? bq : (mat == 1) ? bk : bv;
  _Float16*       Out  = (mat == 0) ? Qh : (mat == 1) ? Kh : Vh;

  __shared__ __align__(16) _Float16 xs[2][64 * 40];   // [m][k]
  __shared__ __align__(16) _Float16 wsT[2][64 * 40];  // [n][k]

  v8f acc[4];
#pragma unroll
  for (int t = 0; t < 4; ++t) {
    float bb = bias[nt + 16 * t + (lane & 15)];
#pragma unroll
    for (int r = 0; r < 8; ++r) acc[t][r] = bb;
  }

  auto stage = [&](int buf, int k0) {
#pragma unroll
    for (int i = 0; i < 16; ++i) {                 // x tile 64x32 -> f16
      int idx = tid + i * 128, rr = idx >> 5, kk = idx & 31;
      xs[buf][rr * 40 + kk] = (_Float16)x[(size_t)(m0 + rr) * D_MODEL + k0 + kk];
    }
#pragma unroll
    for (int i = 0; i < 16; ++i) {                 // W tile 32x64 -> [n][k]
      int idx = tid + i * 128, kk = idx >> 6, nn = idx & 63;
      wsT[buf][nn * 40 + kk] = (_Float16)W[(size_t)(k0 + kk) * D_MODEL + nt + nn];
    }
  };

  stage(0, 0);
  __syncthreads();
  for (int k0 = 0; k0 < D_MODEL; k0 += 32) {
    const int cur = (k0 >> 5) & 1;
    if (k0 + 32 < D_MODEL) stage(cur ^ 1, k0 + 32);
    v16h a  = load_frag_a_h(xs[cur], wave * 16, 40, 0);
    v16h b0 = load_frag_bt(wsT[cur],  0, 40, 0);
    v16h b1 = load_frag_bt(wsT[cur], 16, 40, 0);
    v16h b2 = load_frag_bt(wsT[cur], 32, 40, 0);
    v16h b3 = load_frag_bt(wsT[cur], 48, 40, 0);
    acc[0] = wmma_f16(a, b0, acc[0]);
    acc[1] = wmma_f16(a, b1, acc[1]);
    acc[2] = wmma_f16(a, b2, acc[2]);
    acc[3] = wmma_f16(a, b3, acc[3]);
    __syncthreads();
  }

  const int b = m0 >> 11, s0 = m0 & (SEQ - 1), h = nt >> 6;
#pragma unroll
  for (int t = 0; t < 4; ++t) {
    int d = 16 * t + (lane & 15);
#pragma unroll
    for (int r = 0; r < 8; ++r) {
      int s = s0 + wave * 16 + r + ((lane & 16) ? 8 : 0);
      Out[(((size_t)b * NUM_HEADS + h) * SEQ + s) * DK + d] = (_Float16)acc[t][r];
    }
  }
}

// ---------------------------------------------------------------------------
// Kernel 2: causal flash attention.  grid (S/64=32, B*H=32), block 128.
// Ping-pong tiles: TDM prefetches the NEXT K tile while WMMAs consume the
// current one; V tile staged transposed; tensorcnt drained before barrier.
__global__ __launch_bounds__(128) void attn_kernel(
    const _Float16* __restrict__ Qh, const _Float16* __restrict__ Kh,
    const _Float16* __restrict__ Vh, _Float16* __restrict__ Oh) {
  const int tid = threadIdx.x, lane = tid & 31, wave = tid >> 5;
  const int bh = blockIdx.y;
  const int q0 = blockIdx.x * 64 + wave * 16;
  const _Float16* Qp = Qh + (size_t)bh * SEQ * DK;
  const _Float16* Kp = Kh + (size_t)bh * SEQ * DK;
  const _Float16* Vp = Vh + (size_t)bh * SEQ * DK;

  __shared__ __align__(16) _Float16 Ks[2][32 * 72];   // [key][d], TDM-padded
  __shared__ __align__(16) _Float16 VsT[2][64 * 40];  // [d][key]
  __shared__ __align__(16) _Float16 Ps[4][16 * 40];

  const v16h qa0 = load_frag_a_h(Qp, q0, DK, 0);
  const v16h qa1 = load_frag_a_h(Qp, q0, DK, 32);

  v8f o[4];
  float M[8], L[8];
#pragma unroll
  for (int t = 0; t < 4; ++t)
#pragma unroll
    for (int r = 0; r < 8; ++r) o[t][r] = 0.0f;
#pragma unroll
  for (int r = 0; r < 8; ++r) { M[r] = -1e30f; L[r] = 0.0f; }

  auto tdm_k = [&](int buf, int k0) {
    // 32 rows x 64 f16; pad 4 DWORDs after each 32-DWORD (128B) row
    // -> landed LDS row stride = 144B = 72 halfs.
    tdm_load_2d_f16((unsigned)(unsigned long long)(&Ks[buf][0]),
                    Kp + (size_t)k0 * DK,
                    /*tensor*/ DK, SEQ, /*tile*/ DK, 32,
                    /*stride0*/ DK, /*pad codes*/ 4u, 3u, 1u);
  };
  auto stage_v = [&](int buf, int k0) {
#pragma unroll
    for (int i = 0; i < 16; ++i) {                 // V tile 32x64 -> [d][key]
      int idx = tid + i * 128, rr = idx >> 6, dd = idx & 63;
      VsT[buf][dd * 40 + rr] = Vp[(size_t)(k0 + rr) * DK + dd];
    }
  };

  const float scale = 0.125f;                      // 1/sqrt(64)
  const int nkb = 2 * blockIdx.x + 2;              // causal key-block bound

  if (wave == 0) { tdm_k(0, 0); __builtin_amdgcn_s_wait_tensorcnt(0); }
  stage_v(0, 0);
  __syncthreads();

  for (int kb = 0; kb < nkb; ++kb) {
    const int k0 = kb * 32, cur = kb & 1;
    const bool have_next = (kb + 1 < nkb);
    if (have_next) {                               // overlap next tile with compute
      if (wave == 0) tdm_k(cur ^ 1, k0 + 32);
      stage_v(cur ^ 1, k0 + 32);
    }

    if (k0 <= q0 + 15) {                           // wave-uniform causal skip
      const _Float16* ks = &Ks[cur][0];
      v16h kb0 = load_frag_bt(ks,  0, 72, 0);
      v16h kb1 = load_frag_bt(ks,  0, 72, 32);
      v16h kb2 = load_frag_bt(ks, 16, 72, 0);
      v16h kb3 = load_frag_bt(ks, 16, 72, 32);
      v8f c0 = {}, c1 = {};                        // scores, keys [0,16) / [16,32)
      c0 = wmma_f16(qa0, kb0, c0);
      c1 = wmma_f16(qa0, kb2, c1);
      c0 = wmma_f16(qa1, kb1, c0);
      c1 = wmma_f16(qa1, kb3, c1);

      const int qrb = q0 + ((lane & 16) ? 8 : 0);
      const int key = k0 + (lane & 15);
#pragma unroll
      for (int r = 0; r < 8; ++r) {
        int qr = qrb + r;
        float s0v = c0[r] * scale; if (key > qr)      s0v = -1e30f;
        float s1v = c1[r] * scale; if (key + 16 > qr) s1v = -1e30f;
        float m = fmaxf(s0v, s1v);
#pragma unroll
        for (int off = 8; off >= 1; off >>= 1) m = fmaxf(m, __shfl_xor(m, off, 32));
        float mn = fmaxf(M[r], m);
        float sc = __expf(M[r] - mn);
        float p0 = __expf(s0v - mn), p1 = __expf(s1v - mn);
        float rs = p0 + p1;
#pragma unroll
        for (int off = 8; off >= 1; off >>= 1) rs += __shfl_xor(rs, off, 32);
        L[r] = L[r] * sc + rs;
        M[r] = mn;
#pragma unroll
        for (int t = 0; t < 4; ++t) o[t][r] *= sc;
        c0[r] = p0; c1[r] = p1;
      }

      // C-layout -> A-layout via wave-private LDS patch
      _Float16* ps = &Ps[wave][0];
#pragma unroll
      for (int r = 0; r < 8; ++r) {
        int row = r + ((lane & 16) ? 8 : 0);
        ps[row * 40 + (lane & 15)]      = (_Float16)c0[r];
        ps[row * 40 + 16 + (lane & 15)] = (_Float16)c1[r];
      }
      asm volatile("s_wait_dscnt 0x0" ::: "memory");
      v16h pa = load_frag_a_h(ps, 0, 40, 0);       // P 16x32
      const _Float16* vt = &VsT[cur][0];
      v16h vb0 = load_frag_bt(vt,  0, 40, 0);
      v16h vb1 = load_frag_bt(vt, 16, 40, 0);
      v16h vb2 = load_frag_bt(vt, 32, 40, 0);
      v16h vb3 = load_frag_bt(vt, 48, 40, 0);
      o[0] = wmma_f16(pa, vb0, o[0]);
      o[1] = wmma_f16(pa, vb1, o[1]);
      o[2] = wmma_f16(pa, vb2, o[2]);
      o[3] = wmma_f16(pa, vb3, o[3]);
    }
    if (have_next && wave == 0) __builtin_amdgcn_s_wait_tensorcnt(0);
    __syncthreads();
  }

  const int b = bh >> 4, h = bh & 15;
#pragma unroll
  for (int t = 0; t < 4; ++t) {
    int d = h * DK + 16 * t + (lane & 15);
#pragma unroll
    for (int r = 0; r < 8; ++r) {
      int qr = q0 + r + ((lane & 16) ? 8 : 0);
      Oh[((size_t)b * SEQ + qr) * D_MODEL + d] = (_Float16)(o[t][r] / L[r]);
    }
  }
}

// ---------------------------------------------------------------------------
// Kernel 3: output projection.  out = Oh(f16) @ Wo + bo  (f32 result).
__global__ __launch_bounds__(128) void out_proj_kernel(
    const _Float16* __restrict__ Oh, const float* __restrict__ Wo,
    const float* __restrict__ bo, float* __restrict__ out) {
  const int tid = threadIdx.x, lane = tid & 31, wave = tid >> 5;
  const int m0 = blockIdx.x * 64, nt = blockIdx.y * 64;

  __shared__ __align__(16) _Float16 wsT[2][64 * 40];  // [n][k]

  v8f acc[4];
#pragma unroll
  for (int t = 0; t < 4; ++t) {
    float bb = bo[nt + 16 * t + (lane & 15)];
#pragma unroll
    for (int r = 0; r < 8; ++r) acc[t][r] = bb;
  }

  auto stage = [&](int buf, int k0) {
#pragma unroll
    for (int i = 0; i < 16; ++i) {
      int idx = tid + i * 128, kk = idx >> 6, nn = idx & 63;
      wsT[buf][nn * 40 + kk] = (_Float16)Wo[(size_t)(k0 + kk) * D_MODEL + nt + nn];
    }
  };

  stage(0, 0);
  __syncthreads();
  for (int k0 = 0; k0 < D_MODEL; k0 += 32) {
    const int cur = (k0 >> 5) & 1;
    if (k0 + 32 < D_MODEL) stage(cur ^ 1, k0 + 32);
    v16h a  = load_frag_a_h(Oh, m0 + wave * 16, D_MODEL, k0);  // direct global f16
    v16h b0 = load_frag_bt(wsT[cur],  0, 40, 0);
    v16h b1 = load_frag_bt(wsT[cur], 16, 40, 0);
    v16h b2 = load_frag_bt(wsT[cur], 32, 40, 0);
    v16h b3 = load_frag_bt(wsT[cur], 48, 40, 0);
    acc[0] = wmma_f16(a, b0, acc[0]);
    acc[1] = wmma_f16(a, b1, acc[1]);
    acc[2] = wmma_f16(a, b2, acc[2]);
    acc[3] = wmma_f16(a, b3, acc[3]);
    __syncthreads();
  }

#pragma unroll
  for (int t = 0; t < 4; ++t) {
#pragma unroll
    for (int r = 0; r < 8; ++r) {
      int row = m0 + wave * 16 + r + ((lane & 16) ? 8 : 0);
      out[(size_t)row * D_MODEL + nt + 16 * t + (lane & 15)] = acc[t][r];
    }
  }
}

// ---------------------------------------------------------------------------
extern "C" void kernel_launch(void* const* d_in, const int* in_sizes, int n_in,
                              void* d_out, int out_size, void* d_ws, size_t ws_size,
                              hipStream_t stream) {
  const float* x  = (const float*)d_in[0];
  const float* Wq = (const float*)d_in[1];
  const float* bq = (const float*)d_in[2];
  const float* Wk = (const float*)d_in[3];
  const float* bk = (const float*)d_in[4];
  const float* Wv = (const float*)d_in[5];
  const float* bv = (const float*)d_in[6];
  const float* Wo = (const float*)d_in[7];
  const float* bo = (const float*)d_in[8];

  const size_t nQKV = (size_t)BATCH * NUM_HEADS * SEQ * DK;   // 4M halfs each
  _Float16* Qh = (_Float16*)d_ws;
  _Float16* Kh = Qh + nQKV;
  _Float16* Vh = Kh + nQKV;
  _Float16* Oh = Vh + nQKV;                                   // [B,S,D_MODEL]

  qkv_proj_kernel<<<dim3(64, 48), 128, 0, stream>>>(x, Wq, bq, Wk, bk, Wv, bv,
                                                    Qh, Kh, Vh);
  attn_kernel<<<dim3(32, 32), 128, 0, stream>>>(Qh, Kh, Vh, Oh);
  out_proj_kernel<<<dim3(64, 16), 128, 0, stream>>>(Oh, Wo, bo, (float*)d_out);
}
